// EmbeddingLoss_15092515078409
// MI455X (gfx1250) — compile-verified, matrix operand order checked by reading, not run on hardware
//
#include <hip/hip_runtime.h>
#include <stdint.h>

typedef __attribute__((ext_vector_type(16))) __bf16 v16bf;
typedef __attribute__((ext_vector_type(8)))  float  v8f;

#define HW        (512*512)
#define DCH       64
#define NW        20
#define NC        27
#define NTOT      48          // 47 label rows padded to 48 (3 N-tiles of 16)
#define TPB       256
#define PPB       256         // pixels per block
#define AP_STRIDE 36          // u32 row stride for packed bf16 A (pad vs bank conflicts, 16B-aligned rows)
#define LG_STRIDE 49          // float row stride for logits (gcd(49,64)=1 -> conflict-free column reads)

__device__ __forceinline__ uint32_t f2bf(float f) {
    uint32_t u = __float_as_uint(f);
    return (u + 0x7FFFu + ((u >> 16) & 1u)) >> 16;   // RNE truncate to bf16
}

__global__ void zero_out_kernel(float* out, int n) {
    int i = blockIdx.x * blockDim.x + threadIdx.x;
    if (i < n) out[i] = 0.0f;
}

// Build combined bf16 weight matrix [48][64]: rows 0..19 = w_within, 20..46 = w_cross (T=1), 47 = 0
__global__ void stage_weights_kernel(const float* __restrict__ ww,
                                     const float* __restrict__ wc,
                                     uint16_t* __restrict__ wcomb) {
    int i = blockIdx.x * blockDim.x + threadIdx.x;
    if (i >= NTOT * DCH) return;
    int n = i / DCH, d = i % DCH;
    float v = 0.0f;
    if (n < NW)            v = ww[n * DCH + d];
    else if (n < NW + NC)  v = wc[(n - NW) * DCH + d];
    wcomb[i] = (uint16_t)f2bf(v);
}

__global__ __launch_bounds__(TPB)
void fused_entropy_centroid_kernel(const float* __restrict__ enc,
                                   const int*   __restrict__ tgt,
                                   const uint16_t* __restrict__ wcomb,
                                   float* __restrict__ out) {
    // A-staging and logits share the same LDS (phases separated by barriers)
    __shared__ __align__(16) union {
        uint32_t a[PPB * AP_STRIDE];      // packed bf16 pairs: [pixel][kpair]
        float    lg[PPB * LG_STRIDE];     // logits: [pixel][label]
    } sh;
    __shared__ float cent[NW * 65];       // padded rows -> labels land in distinct banks
    __shared__ int   cnt[NW];
    __shared__ float entsum;

    const int t    = threadIdx.x;
    const int lane = t & 31;
    const int wave = t >> 5;

    for (int i = t; i < NW * 65; i += TPB) cent[i] = 0.0f;
    if (t < NW) cnt[t] = 0;
    if (t == 0) entsum = 0.0f;
    __syncthreads();

    const long long pb    = (long long)blockIdx.x * PPB;
    const int       bimg  = (int)(pb / HW);
    const int       hw    = (int)(pb % HW);
    const long long gbase = (long long)bimg * DCH * HW + hw;

    const int lab = tgt[pb + t];
    atomicAdd(&cnt[lab], 1);

    // ---- B fragments: gather per-lane from staged bf16 weights (tiny, cache-resident)
    // 16-bit operand layout: lane -> (col = lane&15, half = lane>>4);
    // VGPR v holds packed K-pair k_even = (v<4 ? 2v : 2(v-4)+16) + 8*half + 32*khalf
    v16bf Bfrag[3][2];
    {
        const int nloc = lane & 15;
        const int half = lane >> 4;
        for (int nt = 0; nt < 3; ++nt) {
            const int n = nt * 16 + nloc;
            for (int kh = 0; kh < 2; ++kh) {
                union { uint32_t u[8]; v16bf v; } bu;
                #pragma unroll
                for (int v = 0; v < 8; ++v) {
                    int ke = ((v < 4) ? 2 * v : 2 * (v - 4) + 16) + 8 * half + 32 * kh;
                    uint32_t lo = wcomb[n * DCH + ke];
                    uint32_t hi = wcomb[n * DCH + ke + 1];
                    bu.u[v] = lo | (hi << 16);
                }
                Bfrag[nt][kh] = bu.v;
            }
        }
    }

    // ---- single pass over encoder tile: coalesced loads feed BOTH the bf16 A-staging
    //      (packed channel pairs) and the fp32 centroid LDS accumulation.
    for (int i = 0; i < DCH / 2; ++i) {
        float fa = enc[gbase + (long long)(2 * i)     * HW + t];
        float fb = enc[gbase + (long long)(2 * i + 1) * HW + t];
        sh.a[t * AP_STRIDE + i] = f2bf(fa) | (f2bf(fb) << 16);
        atomicAdd(&cent[lab * 65 + 2 * i],     fa);
        atomicAdd(&cent[lab * 65 + 2 * i + 1], fb);
    }
    __syncthreads();

    // ---- load A fragments for this wave's 2 M-tiles (2x ds_load_b128 per K-half)
    union { uint32_t u[8]; v16bf v; } Af[2][2];   // [mtile][khalf]
    {
        const int mrow = (lane & 15);
        const int c0   = 4 * (lane >> 4);
        #pragma unroll
        for (int mt = 0; mt < 2; ++mt) {
            const int m = (wave * 2 + mt) * 16 + mrow;
            #pragma unroll
            for (int kh = 0; kh < 2; ++kh) {
                *(uint4*)&Af[mt][kh].u[0] = *(const uint4*)&sh.a[m * AP_STRIDE + c0 + 16 * kh];
                *(uint4*)&Af[mt][kh].u[4] = *(const uint4*)&sh.a[m * AP_STRIDE + c0 + 16 * kh + 8];
            }
        }
    }
    __syncthreads();   // all waves have A in registers; LDS can be reused for logits

    // ---- WMMA: per M-tile, 3 N-tiles, K=64 via 2 chained 16x16x32 bf16 WMMAs
    #pragma unroll
    for (int mt = 0; mt < 2; ++mt) {
        const int mbase = (wave * 2 + mt) * 16 + 8 * (lane >> 4);
        #pragma unroll
        for (int nt = 0; nt < 3; ++nt) {
            v8f acc = {};
            acc = __builtin_amdgcn_wmma_f32_16x16x32_bf16(
                false, Af[mt][0].v, false, Bfrag[nt][0], (short)0, acc, false, false);
            acc = __builtin_amdgcn_wmma_f32_16x16x32_bf16(
                false, Af[mt][1].v, false, Bfrag[nt][1], (short)0, acc, false, false);
            // D layout: lane col n = nt*16 + (lane&15); VGPR v -> row mbase + v
            const int n = nt * 16 + (lane & 15);
            #pragma unroll
            for (int v = 0; v < 8; ++v)
                sh.lg[(mbase + v) * LG_STRIDE + n] = acc[v];
        }
    }
    __syncthreads();

    // ---- per-pixel softmax entropies: within = logits[0..18], cross = logits[20..45]
    float l[46];
    #pragma unroll
    for (int j = 0; j < 46; ++j) l[j] = sh.lg[t * LG_STRIDE + j];

    float mw = l[0];
    #pragma unroll
    for (int j = 1; j < 19; ++j) mw = fmaxf(mw, l[j]);
    float se = 0.0f, sxe = 0.0f;
    #pragma unroll
    for (int j = 0; j < 19; ++j) { float s = l[j] - mw; float e = __expf(s); se += e; sxe += s * e; }
    const float Hw = __logf(se) - sxe / se;

    float mc = l[20];
    #pragma unroll
    for (int j = 21; j < 46; ++j) mc = fmaxf(mc, l[j]);
    se = 0.0f; sxe = 0.0f;
    #pragma unroll
    for (int j = 20; j < 46; ++j) { float s = l[j] - mc; float e = __expf(s); se += e; sxe += s * e; }
    const float Hc = __logf(se) - sxe / se;

    atomicAdd(&entsum, 0.5f * Hw + 0.5f * Hc);
    __syncthreads();

    // ---- global reductions (tiny: 1 + 1280 + 20 floats)
    if (t == 0) atomicAdd(out, entsum * (1.0f / (float)(4 * HW)));
    for (int i = t; i < NW * DCH; i += TPB) {
        int n = i / DCH, d = i % DCH;
        atomicAdd(out + 1 + i, cent[n * 65 + d]);
    }
    if (t < NW) atomicAdd(out + 1 + NW * DCH + t, (float)cnt[t]);
}

extern "C" void kernel_launch(void* const* d_in, const int* in_sizes, int n_in,
                              void* d_out, int out_size, void* d_ws, size_t ws_size,
                              hipStream_t stream) {
    const float* enc = (const float*)d_in[0];   // (4,64,512,512) fp32
    const float* ww  = (const float*)d_in[1];   // (20,64) fp32
    const float* wc  = (const float*)d_in[2];   // (27,64) fp32
    const int*   tgt = (const int*)d_in[3];     // (4,512,512) int32
    float*       out = (float*)d_out;           // [loss(1) | centroids(1280) | counts(20)]
    uint16_t*  wcomb = (uint16_t*)d_ws;         // staged bf16 combined weights [48][64]

    const int npix = in_sizes[3];               // 4*512*512

    zero_out_kernel<<<(out_size + TPB - 1) / TPB, TPB, 0, stream>>>(out, out_size);
    stage_weights_kernel<<<(NTOT * DCH + TPB - 1) / TPB, TPB, 0, stream>>>(ww, wc, wcomb);
    fused_entropy_centroid_kernel<<<npix / PPB, TPB, 0, stream>>>(enc, tgt, wcomb, out);
}
